// AttentionBlock_70317204570334
// MI455X (gfx1250) — matrix operand order repaired and checked by reading, not compiled, not run on hardware
//
#include <hip/hip_runtime.h>
#include <stdint.h>

#define EMBED   768
#define HIDDEN  3072
#define HEADS   12
#define HEAD_DIM 64
#define BATCH   16
#define SEQ     1024
#define MROWS   (BATCH * SEQ)          // 16384
#define QKV_COLS (3 * EMBED)           // 2304

typedef __attribute__((ext_vector_type(16))) __bf16 v16bf;
typedef __attribute__((ext_vector_type(8)))  float  v8f;
typedef unsigned int v4u __attribute__((ext_vector_type(4)));
typedef int          v4i __attribute__((ext_vector_type(4)));
typedef int          v8i __attribute__((ext_vector_type(8)));

#ifndef __has_builtin
#define __has_builtin(x) 0
#endif
#if defined(__AMDGCN__) && __has_builtin(__builtin_amdgcn_tensor_load_to_lds)
#define HAVE_TDM 1
#else
#define HAVE_TDM 0
#endif

union FragU { v16bf v; uint4 u[2]; };

__device__ __forceinline__ unsigned short f32_to_bf16(float f) {
  union { float f; unsigned u; } c; c.f = f;
  unsigned u = c.u;
  return (unsigned short)((u + 0x7FFFu + ((u >> 16) & 1u)) >> 16);
}

// Load a 16-element bf16 fragment as two 16-byte chunks (chunk index in 16B units)
__device__ __forceinline__ v16bf frag_ld(const unsigned short* base16, int c0, int c1) {
  const uint4* p = (const uint4*)base16;
  FragU f; f.u[0] = p[c0]; f.u[1] = p[c1];
  return f.v;
}

__device__ __forceinline__ v8f wmma_bf16(v16bf a, v16bf b, v8f c) {
  return __builtin_amdgcn_wmma_f32_16x16x32_bf16(false, a, false, b, (short)0, c, false, false);
}

// ---------------------------------------------------------------- TDM tile load
// DMA one [128 rows x 32 cols] bf16 tile (row stride K elements in global) into
// LDS at lds_off, padding each 64B row with 16B -> LDS row stride 80B (= LDST).
#define BM 128
#define BN 128
#define BK 32
#define LDST 40   // LDS row stride in elements (80B, 16B-aligned, conflict-free)

__device__ __forceinline__ void tdm_load_tile(unsigned lds_off,
                                              const unsigned short* gptr,
                                              unsigned K) {
#if HAVE_TDM
  unsigned long long ga = (unsigned long long)(uintptr_t)gptr;
  v4u g0;
  g0[0] = 1u;                                        // count=1, user descriptor
  g0[1] = lds_off;                                   // lds_addr (bytes)
  g0[2] = (unsigned)(ga & 0xFFFFFFFFu);              // global_addr[31:0]
  g0[3] = (unsigned)((ga >> 32) & 0x01FFFFFFu)       // global_addr[56:32]
        | (2u << 30);                                // type = 2 ("image")
  // group1 control word: data_size=2B, pad_enable, pad_interval=64B(code 3),
  // pad_amount=4 DWORDs(code 3)
  const unsigned ctrl = (1u << 16) | (1u << 20) | (3u << 22) | (3u << 25);
  unsigned long long q0 = (unsigned long long)ctrl
                        | ((unsigned long long)(K & 0xFFFFu) << 48);     // tensor_dim0 lo
  unsigned long long q1 = ((unsigned long long)K >> 16)                  // tensor_dim0 hi
                        | (128ull << 16)                                 // tensor_dim1 = 128 rows
                        | (32ull << 48);                                 // tile_dim0 = 32
  unsigned long long q2 = 128ull                                         // tile_dim1 = 128
                        | ((unsigned long long)K << 32);                 // dim0_stride lo32
  unsigned long long q3 = 0ull;                                          // stride hi / dim1_stride
  v8i g1;
  g1[0] = (int)(unsigned)(q0 & 0xFFFFFFFFu);
  g1[1] = (int)(unsigned)(q0 >> 32);
  g1[2] = (int)(unsigned)(q1 & 0xFFFFFFFFu);
  g1[3] = (int)(unsigned)(q1 >> 32);
  g1[4] = (int)(unsigned)(q2 & 0xFFFFFFFFu);
  g1[5] = (int)(unsigned)(q2 >> 32);
  g1[6] = (int)(unsigned)(q3 & 0xFFFFFFFFu);
  g1[7] = (int)(unsigned)(q3 >> 32);
  v4i gz = (v4i){0, 0, 0, 0};
#if __clang_major__ >= 23
  v8i gz8 = (v8i){0, 0, 0, 0, 0, 0, 0, 0};
  __builtin_amdgcn_tensor_load_to_lds(g0, g1, gz, gz, gz8, 0);
#else
  __builtin_amdgcn_tensor_load_to_lds(g0, g1, gz, gz, 0);
#endif
#else
  (void)lds_off; (void)gptr; (void)K;
#endif
}

// ---------------------------------------------------------------- fp32 -> bf16
__global__ void cvt_bf16_kernel(const float* __restrict__ in,
                                unsigned short* __restrict__ out, int n) {
  int i = blockIdx.x * blockDim.x + threadIdx.x;
  if (i < n) out[i] = f32_to_bf16(in[i]);
}

// ---------------------------------------------------------------- LayerNorm (768 wide) -> bf16
__global__ __launch_bounds__(256) void ln_kernel(const float* __restrict__ x,
                                                 const float* __restrict__ g,
                                                 const float* __restrict__ b,
                                                 unsigned short* __restrict__ out) {
  __shared__ float red_s[8];
  __shared__ float red_q[8];
  const int row = blockIdx.x;
  const int t = threadIdx.x;
  const float* xr = x + (size_t)row * EMBED;
  float v0 = xr[t], v1 = xr[t + 256], v2 = xr[t + 512];
  float s = v0 + v1 + v2;
  float q = v0 * v0 + v1 * v1 + v2 * v2;
  #pragma unroll
  for (int m = 1; m < 32; m <<= 1) {
    s += __shfl_xor(s, m, 32);
    q += __shfl_xor(q, m, 32);
  }
  const int wave = t >> 5, lane = t & 31;
  if (lane == 0) { red_s[wave] = s; red_q[wave] = q; }
  __syncthreads();
  s = 0.f; q = 0.f;
  #pragma unroll
  for (int i = 0; i < 8; ++i) { s += red_s[i]; q += red_q[i]; }
  const float mu = s * (1.0f / EMBED);
  const float var = q * (1.0f / EMBED) - mu * mu;
  const float rs = rsqrtf(var + 1e-6f);
  unsigned short* orow = out + (size_t)row * EMBED;
  orow[t]       = f32_to_bf16((v0 - mu) * rs * g[t]       + b[t]);
  orow[t + 256] = f32_to_bf16((v1 - mu) * rs * g[t + 256] + b[t + 256]);
  orow[t + 512] = f32_to_bf16((v2 - mu) * rs * g[t + 512] + b[t + 512]);
}

// ---------------------------------------------------------------- tiled WMMA GEMM
// C[M,Nt] = A[M,K](bf16) * W[Nt,K](bf16)^T + bias
// MODE 0: store bf16    MODE 1: store fp32 (resid + acc + bias)    MODE 2: bf16 gelu(acc+bias)
// TDM path: wave 0 issues tensor_load_to_lds for next tile pair (double-buffered,
// TENSORcnt-pipelined) while all 8 waves run WMMAs on the current tile.
template <int MODE>
__global__ __launch_bounds__(256) void gemm_wmma(const unsigned short* __restrict__ A,
                                                 const unsigned short* __restrict__ W,
                                                 const float* __restrict__ bias,
                                                 const float* __restrict__ resid,
                                                 void* __restrict__ outp,
                                                 int K, int Nt) {
  __shared__ unsigned short a_lds[2][BM * LDST];
  __shared__ unsigned short w_lds[2][BN * LDST];

  const int t    = threadIdx.x;
  const int lane = t & 31;
  const int wave = t >> 5;
  const int wm = wave & 1;      // 2 waves along M (64 rows each)
  const int wn = wave >> 1;     // 4 waves along N (32 cols each)
  const int hi = (lane >> 4) & 1;
  const int ln = lane & 15;
  const int bm = blockIdx.y * BM;
  const int bn = blockIdx.x * BN;
  const int nk = K / BK;

  v8f acc[4][2];
  #pragma unroll
  for (int mi = 0; mi < 4; ++mi)
    #pragma unroll
    for (int ni = 0; ni < 2; ++ni) acc[mi][ni] = (v8f){};

#if HAVE_TDM
  const unsigned a_off0 = (unsigned)(uintptr_t)(void*)&a_lds[0][0];
  const unsigned a_off1 = (unsigned)(uintptr_t)(void*)&a_lds[1][0];
  const unsigned w_off0 = (unsigned)(uintptr_t)(void*)&w_lds[0][0];
  const unsigned w_off1 = (unsigned)(uintptr_t)(void*)&w_lds[1][0];
  const unsigned short* arow = A + (size_t)bm * K;
  const unsigned short* wrow = W + (size_t)bn * K;
  if (t < 32) {  // wave 0 drives the TDM
    tdm_load_tile(a_off0, arow, (unsigned)K);
    tdm_load_tile(w_off0, wrow, (unsigned)K);
  }
#else
  const int lrow = t >> 1;          // 0..127
  const int lch  = (t & 1) * 2;     // chunk pair {0,1} or {2,3}
#endif

  for (int i = 0; i < nk; ++i) {
    const int cur = i & 1;
#if HAVE_TDM
    if (t < 32) {
      if (i + 1 < nk) {
        const unsigned kn = (unsigned)((i + 1) * BK);
        tdm_load_tile(cur ? a_off0 : a_off1, arow + kn, (unsigned)K);
        tdm_load_tile(cur ? w_off0 : w_off1, wrow + kn, (unsigned)K);
        __builtin_amdgcn_s_wait_tensorcnt(2);   // current tile pair resident
      } else {
        __builtin_amdgcn_s_wait_tensorcnt(0);
      }
    }
    __syncthreads();
#else
    const uint4* ga = (const uint4*)(A + (size_t)(bm + lrow) * K + i * BK);
    const uint4* gw = (const uint4*)(W + (size_t)(bn + lrow) * K + i * BK);
    uint4 a0 = ga[lch], a1 = ga[lch + 1];
    uint4 w0 = gw[lch], w1 = gw[lch + 1];
    __syncthreads();
    uint4* sa = (uint4*)(&a_lds[cur][0] + lrow * LDST);
    sa[lch] = a0; sa[lch + 1] = a1;
    uint4* sw = (uint4*)(&w_lds[cur][0] + lrow * LDST);
    sw[lch] = w0; sw[lch + 1] = w1;
    __syncthreads();
#endif

    v16bf af[4], bfr[2];
    #pragma unroll
    for (int mi = 0; mi < 4; ++mi) {
      const int row = wm * 64 + mi * 16 + ln;
      af[mi] = frag_ld(&a_lds[cur][0] + row * LDST, hi ? 1 : 0, hi ? 3 : 2);  // A lane layout
    }
    #pragma unroll
    for (int ni = 0; ni < 2; ++ni) {
      const int row = wn * 32 + ni * 16 + ln;
      bfr[ni] = frag_ld(&w_lds[cur][0] + row * LDST, hi ? 1 : 0, hi ? 3 : 2); // B lane layout
    }
    #pragma unroll
    for (int mi = 0; mi < 4; ++mi)
      #pragma unroll
      for (int ni = 0; ni < 2; ++ni)
        acc[mi][ni] = wmma_bf16(af[mi], bfr[ni], acc[mi][ni]);

    __syncthreads();   // protect buffer i&1 (cur) before next-iteration DMA refills it
  }

  #pragma unroll
  for (int ni = 0; ni < 2; ++ni) {
    const int gcol = bn + wn * 32 + ni * 16 + ln;
    const float bv = bias[gcol];
    #pragma unroll
    for (int mi = 0; mi < 4; ++mi) {
      #pragma unroll
      for (int r = 0; r < 8; ++r) {
        const int grow = bm + wm * 64 + mi * 16 + r + hi * 8;
        const size_t idx = (size_t)grow * Nt + gcol;
        float v = acc[mi][ni][r] + bv;
        if constexpr (MODE == 0) {
          ((unsigned short*)outp)[idx] = f32_to_bf16(v);
        } else if constexpr (MODE == 1) {
          ((float*)outp)[idx] = resid[idx] + v;
        } else {
          float gel = 0.5f * v * (1.0f + erff(v * 0.70710678118654752f));
          ((unsigned short*)outp)[idx] = f32_to_bf16(gel);
        }
      }
    }
  }
}

// ---------------------------------------------------------------- flash attention
// qkv: [B*N, 3*EMBED] bf16 laid out (sel, head, dim);  ctx: [B*N, EMBED] bf16
__global__ __launch_bounds__(128) void attention_kernel(const unsigned short* __restrict__ qkv,
                                                        unsigned short* __restrict__ ctx) {
  __shared__ unsigned short k_lds[32 * 72];       // [key][dim], stride 144B
  __shared__ unsigned short vT_lds[64 * LDST];    // [dim][key], stride 80B
  __shared__ unsigned short p_lds[4][16 * LDST];  // per-wave P scratch

  const int bh = blockIdx.x;
  const int b = bh / HEADS, h = bh % HEADS;
  const int t = threadIdx.x;
  const int lane = t & 31, wave = t >> 5;
  const int hi = (lane >> 4) & 1, ln = lane & 15;
  const int qbase = blockIdx.y * 64 + wave * 16;

  const size_t rs = QKV_COLS;
  const unsigned short* qp = qkv + (size_t)(b * SEQ) * rs + 0 * EMBED + h * HEAD_DIM;
  const unsigned short* kp = qkv + (size_t)(b * SEQ) * rs + 1 * EMBED + h * HEAD_DIM;
  const unsigned short* vp = qkv + (size_t)(b * SEQ) * rs + 2 * EMBED + h * HEAD_DIM;

  // Q fragments (held in registers for the whole pass)
  const uint4* qrp = (const uint4*)(qp + (size_t)(qbase + ln) * rs);
  v16bf qf0, qf1;
  { FragU f; f.u[0] = qrp[hi ? 1 : 0]; f.u[1] = qrp[hi ? 3 : 2]; qf0 = f.v; }  // d 0..31
  { FragU f; f.u[0] = qrp[hi ? 5 : 4]; f.u[1] = qrp[hi ? 7 : 6]; qf1 = f.v; }  // d 32..63

  v8f o0 = (v8f){}, o1 = (v8f){}, o2 = (v8f){}, o3 = (v8f){};
  float mrow[8], lsum[8];
  #pragma unroll
  for (int r = 0; r < 8; ++r) { mrow[r] = -1e30f; lsum[r] = 0.0f; }

  const int keyid = t & 31, dgrp = t >> 5;  // tile-load assignment

  for (int kt = 0; kt < SEQ; kt += 32) {
    // cooperative staging of K (row-major) and V (transposed) tiles
    const uint4* gk = (const uint4*)(kp + (size_t)(kt + keyid) * rs + dgrp * 16);
    const uint4* gv = (const uint4*)(vp + (size_t)(kt + keyid) * rs + dgrp * 16);
    uint4 k0 = gk[0], k1 = gk[1];
    uint4 v0 = gv[0], v1 = gv[1];
    __syncthreads();
    uint4* sk = (uint4*)(k_lds + keyid * 72 + dgrp * 16);
    sk[0] = k0; sk[1] = k1;
    union { uint4 u[2]; unsigned short s[16]; } vv;
    vv.u[0] = v0; vv.u[1] = v1;
    #pragma unroll
    for (int i = 0; i < 16; ++i)
      vT_lds[(dgrp * 16 + i) * LDST + keyid] = vv.s[i];
    __syncthreads();

    // S = Q K^T for this wave's 16 q-rows x 32 keys
    v8f s_lo = (v8f){}, s_hi = (v8f){};
    {
      v16bf kb;
      kb = frag_ld(k_lds + (0  + ln) * 72, hi ? 1 : 0, hi ? 3 : 2); s_lo = wmma_bf16(qf0, kb, s_lo);
      kb = frag_ld(k_lds + (0  + ln) * 72, hi ? 5 : 4, hi ? 7 : 6); s_lo = wmma_bf16(qf1, kb, s_lo);
      kb = frag_ld(k_lds + (16 + ln) * 72, hi ? 1 : 0, hi ? 3 : 2); s_hi = wmma_bf16(qf0, kb, s_hi);
      kb = frag_ld(k_lds + (16 + ln) * 72, hi ? 5 : 4, hi ? 7 : 6); s_hi = wmma_bf16(qf1, kb, s_hi);
    }

    // online softmax (row = half-wave of 16 lanes)
    unsigned short* pw = &p_lds[wave][0];
    #pragma unroll
    for (int r = 0; r < 8; ++r) {
      float slo = s_lo[r] * 0.125f, shi = s_hi[r] * 0.125f;
      float mx = fmaxf(slo, shi);
      mx = fmaxf(mx, __shfl_xor(mx, 1, 32));
      mx = fmaxf(mx, __shfl_xor(mx, 2, 32));
      mx = fmaxf(mx, __shfl_xor(mx, 4, 32));
      mx = fmaxf(mx, __shfl_xor(mx, 8, 32));
      const float mnew = fmaxf(mrow[r], mx);
      const float corr = __expf(mrow[r] - mnew);
      const float plo = __expf(slo - mnew), phi = __expf(shi - mnew);
      float ps = plo + phi;
      ps += __shfl_xor(ps, 1, 32);
      ps += __shfl_xor(ps, 2, 32);
      ps += __shfl_xor(ps, 4, 32);
      ps += __shfl_xor(ps, 8, 32);
      lsum[r] = lsum[r] * corr + ps;
      mrow[r] = mnew;
      o0[r] *= corr; o1[r] *= corr; o2[r] *= corr; o3[r] *= corr;
      const int prow = r + hi * 8;
      pw[prow * LDST + ln]      = f32_to_bf16(plo);
      pw[prow * LDST + 16 + ln] = f32_to_bf16(phi);
    }
    asm volatile("s_wait_dscnt 0" ::: "memory");  // wave-private LDS RAW fence

    // O += P V  (4 output dim-tiles)
    v16bf pf = frag_ld(pw + ln * LDST, hi ? 1 : 0, hi ? 3 : 2);
    v16bf vb;
    vb = frag_ld(vT_lds + (0 * 16 + ln) * LDST, hi ? 1 : 0, hi ? 3 : 2); o0 = wmma_bf16(pf, vb, o0);
    vb = frag_ld(vT_lds + (1 * 16 + ln) * LDST, hi ? 1 : 0, hi ? 3 : 2); o1 = wmma_bf16(pf, vb, o1);
    vb = frag_ld(vT_lds + (2 * 16 + ln) * LDST, hi ? 1 : 0, hi ? 3 : 2); o2 = wmma_bf16(pf, vb, o2);
    vb = frag_ld(vT_lds + (3 * 16 + ln) * LDST, hi ? 1 : 0, hi ? 3 : 2); o3 = wmma_bf16(pf, vb, o3);
  }

  #pragma unroll
  for (int r = 0; r < 8; ++r) {
    const float inv = 1.0f / lsum[r];
    const int qrow = qbase + r + hi * 8;
    const size_t base = (size_t)(b * SEQ + qrow) * EMBED + h * HEAD_DIM;
    ctx[base + 0  + ln] = f32_to_bf16(o0[r] * inv);
    ctx[base + 16 + ln] = f32_to_bf16(o1[r] * inv);
    ctx[base + 32 + ln] = f32_to_bf16(o2[r] * inv);
    ctx[base + 48 + ln] = f32_to_bf16(o3[r] * inv);
  }
}

// ---------------------------------------------------------------- launch
extern "C" void kernel_launch(void* const* d_in, const int* in_sizes, int n_in,
                              void* d_out, int out_size, void* d_ws, size_t ws_size,
                              hipStream_t stream) {
  const float* x      = (const float*)d_in[0];
  const float* ln1_g  = (const float*)d_in[1];
  const float* ln1_b  = (const float*)d_in[2];
  const float* qkv_w  = (const float*)d_in[3];
  const float* qkv_b  = (const float*)d_in[4];
  const float* proj_w = (const float*)d_in[5];
  const float* proj_b = (const float*)d_in[6];
  const float* ln2_g  = (const float*)d_in[7];
  const float* ln2_b  = (const float*)d_in[8];
  const float* fc1_w  = (const float*)d_in[9];
  const float* fc1_b  = (const float*)d_in[10];
  const float* fc2_w  = (const float*)d_in[11];
  const float* fc2_b  = (const float*)d_in[12];
  float* out = (float*)d_out;

  char* ws = (char*)d_ws;
  size_t off = 0;
  auto take = [&](size_t bytes) -> char* {
    char* p = ws + off;
    off += (bytes + 255) & ~(size_t)255;
    return p;
  };
  unsigned short* qkvw_bf  = (unsigned short*)take((size_t)QKV_COLS * EMBED * 2);
  unsigned short* projw_bf = (unsigned short*)take((size_t)EMBED * EMBED * 2);
  unsigned short* fc1w_bf  = (unsigned short*)take((size_t)HIDDEN * EMBED * 2);
  unsigned short* fc2w_bf  = (unsigned short*)take((size_t)EMBED * HIDDEN * 2);
  unsigned short* xn_bf    = (unsigned short*)take((size_t)MROWS * EMBED * 2);   // also ln2 output
  float*          x1       = (float*)take((size_t)MROWS * EMBED * 4);
  unsigned short* qkv_bf   = (unsigned short*)take((size_t)MROWS * QKV_COLS * 2);
  unsigned short* ctx_bf   = (unsigned short*)take((size_t)MROWS * EMBED * 2);
  unsigned short* h1_bf    = qkv_bf;  // overlay qkv+ctx (both dead when h1 is written)

  // weight conversions (fp32 -> bf16)
  {
    int n;
    n = QKV_COLS * EMBED;
    cvt_bf16_kernel<<<(n + 255) / 256, 256, 0, stream>>>(qkv_w, qkvw_bf, n);
    n = EMBED * EMBED;
    cvt_bf16_kernel<<<(n + 255) / 256, 256, 0, stream>>>(proj_w, projw_bf, n);
    n = HIDDEN * EMBED;
    cvt_bf16_kernel<<<(n + 255) / 256, 256, 0, stream>>>(fc1_w, fc1w_bf, n);
    n = EMBED * HIDDEN;
    cvt_bf16_kernel<<<(n + 255) / 256, 256, 0, stream>>>(fc2_w, fc2w_bf, n);
  }

  // LN1 -> xn (bf16)
  ln_kernel<<<MROWS, 256, 0, stream>>>(x, ln1_g, ln1_b, xn_bf);

  // QKV = xn @ qkv_w^T + b   -> bf16
  gemm_wmma<0><<<dim3(QKV_COLS / BN, MROWS / BM), 256, 0, stream>>>(
      xn_bf, qkvw_bf, qkv_b, nullptr, (void*)qkv_bf, EMBED, QKV_COLS);

  // flash attention -> ctx (bf16)
  attention_kernel<<<dim3(BATCH * HEADS, SEQ / 64), 128, 0, stream>>>(qkv_bf, ctx_bf);

  // x1 = x + ctx @ proj_w^T + b  (fp32)
  gemm_wmma<1><<<dim3(EMBED / BN, MROWS / BM), 256, 0, stream>>>(
      ctx_bf, projw_bf, proj_b, x, (void*)x1, EMBED, EMBED);

  // LN2 -> xn (bf16, reuse)
  ln_kernel<<<MROWS, 256, 0, stream>>>(x1, ln2_g, ln2_b, xn_bf);

  // h1 = gelu(xn @ fc1_w^T + b) -> bf16
  gemm_wmma<2><<<dim3(HIDDEN / BN, MROWS / BM), 256, 0, stream>>>(
      xn_bf, fc1w_bf, fc1_b, nullptr, (void*)h1_bf, EMBED, HIDDEN);

  // out = x1 + h1 @ fc2_w^T + b  (fp32)
  gemm_wmma<1><<<dim3(EMBED / BN, MROWS / BM), 256, 0, stream>>>(
      h1_bf, fc2w_bf, fc2_b, x1, (void*)out, HIDDEN, EMBED);

  (void)in_sizes; (void)n_in; (void)out_size; (void)ws_size;
}
